// VectorQuantizer_9388798509766
// MI455X (gfx1250) — matrix operand order repaired and checked by reading, not compiled, not run on hardware
//
#include <hip/hip_runtime.h>
#include <math.h>

typedef __attribute__((ext_vector_type(8)))  __bf16 v8bf;
typedef __attribute__((ext_vector_type(16))) __bf16 v16bf;
typedef __attribute__((ext_vector_type(4)))  float  v4f;
typedef __attribute__((ext_vector_type(8)))  float  v8f;

#define E_DIM     256
#define NUM_E     1024
#define N_VEC     65536                 // 16*64*64 flattened vectors
#define XQ_ELEMS  16777216              // 16*256*64*64
#define ENC_ELEMS 67108864              // N_VEC * NUM_E

#define ROWS_PER_WAVE   64              // 4 row-tiles of 16 per wave
#define WAVES_PER_BLOCK 4
#define TILE_ROWS       256             // rows of x staged in LDS per block
// Row block layout (bf16 elems): [hi 256][pad 8][lo 256]  -> stride 520
//  * lo reachable via DS immediate (+528 B), 16B aligned
//  * stride = 260 dwords = 4 (mod 64)  -> conflict-free wave32 b128 loads
#define ROW_S           520
#define LO_OFF          264

// ---------------------------------------------------------------------------
// bf16 split helpers (RNE, exact bit math)
// ---------------------------------------------------------------------------
static __device__ __forceinline__ unsigned short f32_to_bf16_rne(float f) {
    unsigned u = __builtin_bit_cast(unsigned, f);
    unsigned r = u + 0x7FFFu + ((u >> 16) & 1u);
    return (unsigned short)(r >> 16);
}
static __device__ __forceinline__ float bf16_to_f32(unsigned short h) {
    unsigned u = ((unsigned)h) << 16;
    return __builtin_bit_cast(float, u);
}
static __device__ __forceinline__ v16bf cat8(v8bf lo, v8bf hi) {
    return __builtin_shufflevector(lo, hi, 0,1,2,3,4,5,6,7,8,9,10,11,12,13,14,15);
}
// A fragment (16-bit A 16x32 layout): per-lane K chunks at +0 and +16 elems
static __device__ __forceinline__ v16bf ld_a(const unsigned short* p) {
    return cat8(*(const v8bf*)p, *(const v8bf*)(p + 16));
}
// B fragment: per-lane K chunks at +0 and +16 elems (same as A)
static __device__ __forceinline__ v16bf ld_b(const unsigned short* p) {
    return cat8(*(const v8bf*)p, *(const v8bf*)(p + 16));
}

// ---------------------------------------------------------------------------
// Kernel 1: codebook squared norms (f32) + hi/lo bf16 planes + zero histogram
// ---------------------------------------------------------------------------
__global__ void vq_prep(const float* __restrict__ cb,
                        float* __restrict__ cb_sq,
                        int* __restrict__ counts,
                        unsigned short* __restrict__ cb_h,
                        unsigned short* __restrict__ cb_l) {
    int k = blockIdx.x * blockDim.x + threadIdx.x;
    if (k < NUM_E) {
        const float* row = cb + (size_t)k * E_DIM;
        unsigned short* rh = cb_h + (size_t)k * E_DIM;
        unsigned short* rl = cb_l + (size_t)k * E_DIM;
        float s = 0.0f;
        #pragma unroll 4
        for (int d = 0; d < E_DIM; ++d) {
            float v = row[d];
            s += v * v;
            unsigned short h = f32_to_bf16_rne(v);
            unsigned short l = f32_to_bf16_rne(v - bf16_to_f32(h));
            rh[d] = h;
            rl[d] = l;
        }
        cb_sq[k] = s;
        counts[k] = 0;
    }
}

// ---------------------------------------------------------------------------
// Kernel 2: zero one-hot encodings with b128 stores (enc+2 is 16B aligned)
// ---------------------------------------------------------------------------
__global__ void vq_zero_enc(float* __restrict__ enc) {
    size_t tid = (size_t)blockIdx.x * blockDim.x + threadIdx.x;
    if (tid == 0) {
        enc[0] = 0.0f; enc[1] = 0.0f;
        enc[ENC_ELEMS - 2] = 0.0f; enc[ENC_ELEMS - 1] = 0.0f;
    }
    float4 z = make_float4(0.0f, 0.0f, 0.0f, 0.0f);
    float* base = enc + 2;                       // 16B aligned
    const size_t nq = ((size_t)ENC_ELEMS - 4) / 4;
    const size_t stride = (size_t)gridDim.x * blockDim.x;
    for (size_t q = tid; q < nq; q += stride)
        *(float4*)(base + q * 4) = z;
}

// ---------------------------------------------------------------------------
// Kernel 3: split-bf16 distance GEMM (3 fused passes) + argmin
//   dot ~= xh.eh + xl.eh + xh.el  (error ~1e-6 << reference f32 noise ~3e-5)
//   dist_cmp[k] = |e_k|^2 - 2*dot  (|x|^2 constant per row -> same argmin)
// ---------------------------------------------------------------------------
__global__ __launch_bounds__(WAVES_PER_BLOCK * 32)
void vq_argmin(const float* __restrict__ x,
               const unsigned short* __restrict__ cb_h,
               const unsigned short* __restrict__ cb_l,
               const float* __restrict__ cb_sq,
               int* __restrict__ counts,
               int* __restrict__ idx_out,
               float* __restrict__ enc) {
    extern __shared__ unsigned short xs16[];     // [TILE_ROWS][ROW_S] hi+lo

    const int tid    = threadIdx.x;              // 0..127
    const int n_base = blockIdx.x * TILE_ROWS;
    const int b_img  = n_base >> 12;             // n = b*4096 + s
    const int s_base = n_base & 4095;

    // ---- Phase 1: stage x tile (NCHW -> [row][d]) as interleaved hi/lo ----
    const float* xb = x + (((size_t)b_img * E_DIM) << 12) + s_base;
    #pragma unroll 4
    for (int it = 0; it < 2 * E_DIM; ++it) {
        int row = ((it & 1) << 7) + tid;         // coalesced along s
        int d   = it >> 1;
        float v = xb[((size_t)d << 12) + row];
        unsigned short h = f32_to_bf16_rne(v);
        unsigned short l = f32_to_bf16_rne(v - bf16_to_f32(h));
        xs16[row * ROW_S + d]          = h;
        xs16[row * ROW_S + LO_OFF + d] = l;
    }
    __syncthreads();

    const int lane = tid & 31;
    const int wave = tid >> 5;
    const int half = lane >> 4;                  // A: K chunk select; B: K range
    const int l16  = lane & 15;

    // single B base; all fragments (4 tiles x hi/lo) via DS immediates
    const unsigned short* b0 = xs16 + (wave * ROWS_PER_WAVE + l16) * ROW_S + (half << 3);
    const int a_off = half << 3;

    float best0 = 3.4e38f, best1 = 3.4e38f, best2 = 3.4e38f, best3 = 3.4e38f;
    int   bk0 = 0, bk1 = 0, bk2 = 0, bk3 = 0;

    for (int kt = 0; kt < NUM_E / 16; ++kt) {
        const unsigned short* ch = cb_h + (size_t)(kt * 16 + l16) * E_DIM + a_off;
        const unsigned short* cl = cb_l + (size_t)(kt * 16 + l16) * E_DIM + a_off;
        v8f acc0 = {}, acc1 = {}, acc2 = {}, acc3 = {};
        #pragma unroll 2
        for (int d0 = 0; d0 < E_DIM; d0 += 32) {
            v16bf ah  = ld_a(ch + d0);           // global, L2-hot
            v16bf al  = ld_a(cl + d0);
            v16bf b0h = ld_b(b0 + d0);           // LDS, immediate offsets only
            v16bf b1h = ld_b(b0 + 16 * ROW_S + d0);
            v16bf b2h = ld_b(b0 + 32 * ROW_S + d0);
            v16bf b3h = ld_b(b0 + 48 * ROW_S + d0);
            v16bf b0l = ld_b(b0 + LO_OFF + d0);
            v16bf b1l = ld_b(b0 + 16 * ROW_S + LO_OFF + d0);
            v16bf b2l = ld_b(b0 + 32 * ROW_S + LO_OFF + d0);
            v16bf b3l = ld_b(b0 + 48 * ROW_S + LO_OFF + d0);
            // pass 1: xh . eh   (4 independent chains interleaved)
            acc0 = __builtin_amdgcn_wmma_f32_16x16x32_bf16(false, ah, false, b0h, (short)0, acc0, false, false);
            acc1 = __builtin_amdgcn_wmma_f32_16x16x32_bf16(false, ah, false, b1h, (short)0, acc1, false, false);
            acc2 = __builtin_amdgcn_wmma_f32_16x16x32_bf16(false, ah, false, b2h, (short)0, acc2, false, false);
            acc3 = __builtin_amdgcn_wmma_f32_16x16x32_bf16(false, ah, false, b3h, (short)0, acc3, false, false);
            // pass 2: xh . el
            acc0 = __builtin_amdgcn_wmma_f32_16x16x32_bf16(false, al, false, b0h, (short)0, acc0, false, false);
            acc1 = __builtin_amdgcn_wmma_f32_16x16x32_bf16(false, al, false, b1h, (short)0, acc1, false, false);
            acc2 = __builtin_amdgcn_wmma_f32_16x16x32_bf16(false, al, false, b2h, (short)0, acc2, false, false);
            acc3 = __builtin_amdgcn_wmma_f32_16x16x32_bf16(false, al, false, b3h, (short)0, acc3, false, false);
            // pass 3: xl . eh
            acc0 = __builtin_amdgcn_wmma_f32_16x16x32_bf16(false, ah, false, b0l, (short)0, acc0, false, false);
            acc1 = __builtin_amdgcn_wmma_f32_16x16x32_bf16(false, ah, false, b1l, (short)0, acc1, false, false);
            acc2 = __builtin_amdgcn_wmma_f32_16x16x32_bf16(false, ah, false, b2l, (short)0, acc2, false, false);
            acc3 = __builtin_amdgcn_wmma_f32_16x16x32_bf16(false, ah, false, b3l, (short)0, acc3, false, false);
        }
        // C/D layout: VGPR j holds code M = half*8 + j for column n = l16
        const int kb = kt * 16 + (half << 3);    // 8-aligned -> 32B-aligned loads
        v4f cs_lo = *(const v4f*)(cb_sq + kb);
        v4f cs_hi = *(const v4f*)(cb_sq + kb + 4);
        #pragma unroll
        for (int j = 0; j < 8; ++j) {
            float cs  = (j < 4) ? cs_lo[j & 3] : cs_hi[j & 3];
            float d0v = cs - 2.0f * acc0[j];
            float d1v = cs - 2.0f * acc1[j];
            float d2v = cs - 2.0f * acc2[j];
            float d3v = cs - 2.0f * acc3[j];
            if (d0v < best0) { best0 = d0v; bk0 = kb + j; }
            if (d1v < best1) { best1 = d1v; bk1 = kb + j; }
            if (d2v < best2) { best2 = d2v; bk2 = kb + j; }
            if (d3v < best3) { best3 = d3v; bk3 = kb + j; }
        }
    }

    // merge half-wave code ranges (lane L <-> L+16 share row n), lower-index ties
    float best[4] = { best0, best1, best2, best3 };
    int   bk[4]   = { bk0, bk1, bk2, bk3 };
    #pragma unroll
    for (int t = 0; t < 4; ++t) {
        float ob = __shfl_xor(best[t], 16, 32);
        int   ok = __shfl_xor(bk[t], 16, 32);
        if (ob < best[t] || (ob == best[t] && ok < bk[t])) { best[t] = ob; bk[t] = ok; }
        if (half == 0) {
            int n = n_base + wave * ROWS_PER_WAVE + t * 16 + l16;
            int k = bk[t];
            idx_out[n] = k;
            atomicAdd(counts + k, 1);            // integer: deterministic
            enc[(size_t)n * NUM_E + k] = 1.0f;   // after vq_zero_enc
        }
    }
}

// ---------------------------------------------------------------------------
// Kernel 4: gather codes -> xq (NCHW) + deterministic partial SSE per block
// ---------------------------------------------------------------------------
__global__ void vq_xq_loss(const float* __restrict__ x,
                           const float* __restrict__ cb,
                           const int* __restrict__ idx,
                           float* __restrict__ xq,
                           float* __restrict__ partial) {
    __shared__ float red[256];
    float acc = 0.0f;
    const int nq = XQ_ELEMS / 4;
    const int stride = gridDim.x * blockDim.x;
    for (int qd = blockIdx.x * blockDim.x + threadIdx.x; qd < nq; qd += stride) {
        int i   = qd << 2;                 // 4 consecutive s within one (b,d) row
        int b   = i >> 20;
        int rem = i & 0xFFFFF;
        int d   = rem >> 12;
        int s   = rem & 0xFFF;
        int n   = (b << 12) | s;
        v4f xv = *(const v4f*)(x + i);     // 16B aligned
        int k0 = idx[n + 0], k1 = idx[n + 1], k2 = idx[n + 2], k3 = idx[n + 3];
        float q0 = cb[(size_t)k0 * E_DIM + d];
        float q1 = cb[(size_t)k1 * E_DIM + d];
        float q2 = cb[(size_t)k2 * E_DIM + d];
        float q3 = cb[(size_t)k3 * E_DIM + d];
        xq[i + 0] = q0;                    // xq base only 4B-aligned -> b32 stores
        xq[i + 1] = q1;
        xq[i + 2] = q2;
        xq[i + 3] = q3;
        float e0 = q0 - xv.x, e1 = q1 - xv.y, e2 = q2 - xv.z, e3 = q3 - xv.w;
        acc += e0 * e0 + e1 * e1 + e2 * e2 + e3 * e3;
    }
    red[threadIdx.x] = acc;
    __syncthreads();
    for (int off = 128; off > 0; off >>= 1) {
        if (threadIdx.x < off) red[threadIdx.x] += red[threadIdx.x + off];
        __syncthreads();
    }
    if (threadIdx.x == 0) partial[blockIdx.x] = red[0];
}

// ---------------------------------------------------------------------------
// Kernel 5: final loss + perplexity (single block, deterministic)
// ---------------------------------------------------------------------------
__global__ void vq_finalize(const float* __restrict__ partial,
                            const int* __restrict__ counts,
                            float* __restrict__ loss_out,
                            float* __restrict__ perp_out) {
    __shared__ float red[256];
    const int t = threadIdx.x;

    float s = 0.0f;
    for (int j = t; j < 1024; j += 256) s += partial[j];
    red[t] = s;
    __syncthreads();
    for (int off = 128; off > 0; off >>= 1) {
        if (t < off) red[t] += red[t + off];
        __syncthreads();
    }
    float sum_sq = red[0];
    __syncthreads();

    float s2 = 0.0f;
    for (int j = t; j < 1024; j += 256) {
        float e = (float)counts[j] * (1.0f / 65536.0f);
        s2 += e * logf(e + 1.0e10f);       // faithful reproduction of +1e10
    }
    red[t] = s2;
    __syncthreads();
    for (int off = 128; off > 0; off >>= 1) {
        if (t < off) red[t] += red[t + off];
        __syncthreads();
    }
    if (t == 0) {
        *loss_out = 1.25f * sum_sq / (float)XQ_ELEMS;   // (1+beta)*mse
        *perp_out = expf(-red[0]);
    }
}

// ---------------------------------------------------------------------------
extern "C" void kernel_launch(void* const* d_in, const int* in_sizes, int n_in,
                              void* d_out, int out_size, void* d_ws, size_t ws_size,
                              hipStream_t stream) {
    (void)in_sizes; (void)n_in; (void)out_size; (void)ws_size;
    const float* x  = (const float*)d_in[0];     // [16,256,64,64] f32
    const float* cb = (const float*)d_in[1];     // [1024,256] f32

    float* out  = (float*)d_out;
    float* loss = out;                           // [0]
    float* xq   = out + 1;                       // [1 .. 1+XQ)
    float* perp = out + 1 + XQ_ELEMS;            // [1+XQ]
    float* enc  = out + 2 + XQ_ELEMS;            // [2+XQ .. +ENC)

    char*           ws      = (char*)d_ws;
    float*          cb_sq   = (float*)ws;                        // 1024 f32
    int*            counts  = (int*)(ws + 4096);                 // 1024 i32
    int*            idx     = (int*)(ws + 8192);                 // 65536 i32
    float*          partial = (float*)(ws + 270336);             // 1024 f32
    unsigned short* cb_h    = (unsigned short*)(ws + 274432);    // 1024x256 bf16
    unsigned short* cb_l    = (unsigned short*)(ws + 798720);    // 1024x256 bf16

    vq_prep<<<4, 256, 0, stream>>>(cb, cb_sq, counts, cb_h, cb_l);
    vq_zero_enc<<<8192, 256, 0, stream>>>(enc);

    size_t lds_bytes = (size_t)TILE_ROWS * ROW_S * sizeof(unsigned short); // 266,240 B
    vq_argmin<<<N_VEC / TILE_ROWS, WAVES_PER_BLOCK * 32, lds_bytes, stream>>>(
        x, cb_h, cb_l, cb_sq, counts, idx, enc);

    vq_xq_loss<<<1024, 256, 0, stream>>>(x, cb, idx, xq, partial);
    vq_finalize<<<1, 256, 0, stream>>>(partial, counts, loss, perp);
}
